// ARMFeedForward_19043884990637
// MI455X (gfx1250) — matrix operand order repaired and checked by reading, not compiled, not run on hardware
//
#include <hip/hip_runtime.h>
#include <hip/hip_bf16.h>
#include <math.h>

// ---------------- problem constants ----------------
#define T_TOKENS   8192          // B*S = 4*2048
#define DM         1024          // D_MODEL
#define DE         512           // D_EXPERT
#define NE         8             // N_EXPERTS

typedef __bf16 bf16_t;
typedef __attribute__((ext_vector_type(16))) __bf16 v16bf;
typedef __attribute__((ext_vector_type(8)))  float  v8f;
typedef __attribute__((ext_vector_type(4)))  unsigned int u32x4;
typedef __attribute__((ext_vector_type(8)))  int i32x8;
typedef __attribute__((ext_vector_type(4)))  int i32x4;

union FragU { uint4 q[2]; v16bf v; };

#if __has_builtin(__builtin_amdgcn_tensor_load_to_lds)
#define HAVE_TDM 1
#else
#define HAVE_TDM 0
#endif

// x tile in LDS: TDM pads 4 DWORDs (16B / 8 elems) after every 256 DWORDs (1024B).
// Row r, elem k lives at byte r*2080 + 2k + 16*(k>=512)  ->  row stride 1040 bf16.
#define XS_ROW     1040
// h tile: manual stores, padded stride
#define HS_STRIDE  520           // 512 + 8 -> 1040 B row

// ---------------- workspace layout (bytes) ----------------
constexpr size_t WS_COUNTS = 0;
constexpr size_t WS_TOK    = 256;
constexpr size_t WS_GATE   = WS_TOK  + (size_t)NE * T_TOKENS * 4;
constexpr size_t WS_CINV   = WS_GATE + (size_t)NE * T_TOKENS * 4;
constexpr size_t WS_XB     = 524800;
constexpr size_t WS_W1T    = WS_XB  + (size_t)T_TOKENS * DM * 2;
constexpr size_t WS_W2T    = WS_W1T + (size_t)NE * DE * DM * 2;

// ---------------- helpers ----------------
__device__ __forceinline__ float wred(float v) {
#pragma unroll
  for (int o = 16; o > 0; o >>= 1) v += __shfl_xor(v, o, 32);
  return v;
}

// ---------------- kernels ----------------
__global__ __launch_bounds__(256)
void init_kernel(float* __restrict__ out, int n, int* __restrict__ counts) {
  int i = blockIdx.x * 256 + threadIdx.x;
  if (i < n) out[i] = 0.0f;
  if (i < NE) counts[i] = 0;
}

__global__ __launch_bounds__(256)
void prep_x_kernel(const float* __restrict__ x, bf16_t* __restrict__ xb, int n) {
  int i = blockIdx.x * 256 + threadIdx.x;
  if (i < n) xb[i] = (bf16_t)x[i];
}

// w1 (E, DM, DE) -> w1t (E, DE, DM)
__global__ __launch_bounds__(256)
void prep_w1_kernel(const float* __restrict__ w1, bf16_t* __restrict__ w1t) {
  int i = blockIdx.x * 256 + threadIdx.x;     // i = (e*DE + n)*DM + k
  int k = i & (DM - 1);
  int r = i >> 10;
  int n = r & (DE - 1);
  int e = r >> 9;
  w1t[i] = (bf16_t)w1[((size_t)e * DM + k) * DE + n];
}

// w2 (E, DE, DM) -> w2t (E, DM, DE)
__global__ __launch_bounds__(256)
void prep_w2_kernel(const float* __restrict__ w2, bf16_t* __restrict__ w2t) {
  int i = blockIdx.x * 256 + threadIdx.x;     // i = (e*DM + n)*DE + k
  int k = i & (DE - 1);
  int r = i >> 9;
  int n = r & (DM - 1);
  int e = r >> 10;
  w2t[i] = (bf16_t)w2[((size_t)e * DE + k) * DM + n];
}

__global__ __launch_bounds__(256)
void cnorm_kernel(const float* __restrict__ cent, float* __restrict__ cinv) {
  int lane = threadIdx.x & 31;
  int e    = threadIdx.x >> 5;
  float s = 0.0f;
#pragma unroll
  for (int i = 0; i < DM / 32; ++i) {
    float v = cent[e * DM + lane + 32 * i];
    s += v * v;
  }
  s = wred(s);
  if (lane == 0) cinv[e] = 1.0f / fmaxf(sqrtf(s), 1e-12f);
}

__global__ __launch_bounds__(256)
void route_kernel(const float* __restrict__ x, const float* __restrict__ cent,
                  const float* __restrict__ wr, const float* __restrict__ cinv,
                  int* __restrict__ counts, int* __restrict__ tok_list,
                  float* __restrict__ gate_list) {
  int lane = threadIdx.x & 31;
  int t = blockIdx.x * 8 + (threadIdx.x >> 5);
  const float* xr = x + (size_t)t * DM;

  float xv[DM / 32];
  float xx = 0.0f;
#pragma unroll
  for (int i = 0; i < DM / 32; ++i) {
    xv[i] = xr[lane + 32 * i];
    xx += xv[i] * xv[i];
  }
  xx = wred(xx);
  float inv_xn = 1.0f / fmaxf(sqrtf(xx), 1e-12f);

  float l[NE];
#pragma unroll
  for (int e = 0; e < NE; ++e) {
    const float* ce = cent + e * DM;
    const float* we = wr + e * DM;
    float dc = 0.0f, dr = 0.0f;
#pragma unroll
    for (int i = 0; i < DM / 32; ++i) {
      int d = lane + 32 * i;
      dc += xv[i] * ce[d];
      dr += xv[i] * we[d];
    }
    dc = wred(dc);
    dr = wred(dr);
    l[e] = dc * inv_xn * cinv[e] + dr;
  }

  int i1 = 0; float l1 = l[0];
#pragma unroll
  for (int e = 1; e < NE; ++e) if (l[e] > l1) { l1 = l[e]; i1 = e; }
  int i2 = (i1 == 0) ? 1 : 0; float l2 = l[i2];
#pragma unroll
  for (int e = 0; e < NE; ++e) if (e != i1 && l[e] > l2) { l2 = l[e]; i2 = e; }
  float g1 = 1.0f / (1.0f + expf(l2 - l1));
  float g2 = 1.0f - g1;

  if (lane == 0) {
    int s1 = atomicAdd(counts + i1, 1);
    tok_list[i1 * T_TOKENS + s1]  = t;
    gate_list[i1 * T_TOKENS + s1] = g1;
    int s2 = atomicAdd(counts + i2, 1);
    tok_list[i2 * T_TOKENS + s2]  = t;
    gate_list[i2 * T_TOKENS + s2] = g2;
  }
}

// Fused expert FFN over a 16-token tile of one expert's gathered list.
__global__ __launch_bounds__(256)
void moe_ffn_kernel(const bf16_t* __restrict__ xb,
                    const bf16_t* __restrict__ w1t,
                    const bf16_t* __restrict__ w2t,
                    const float* __restrict__ b1,
                    const float* __restrict__ b2,
                    const int* __restrict__ counts,
                    const int* __restrict__ tok_list,
                    const float* __restrict__ gate_list,
                    float* __restrict__ out) {
  const int e    = blockIdx.y;
  const int tile = blockIdx.x;
  const int count = counts[e];
  const int base  = tile * 16;
  if (base >= count) return;   // uniform early-exit, before any barrier

  __shared__ __align__(16) bf16_t xs[16 * XS_ROW];
  __shared__ __align__(16) bf16_t hs[16 * HS_STRIDE];
  __shared__ int   tok[16];
  __shared__ float gv[16];

  const int tid  = threadIdx.x;
  const int lane = tid & 31;
  const int wid  = tid >> 5;

  if (tid < 16) {
    int r = base + tid;
    bool valid = r < count;
    tok[tid] = valid ? tok_list[e * T_TOKENS + r] : 0;
    gv[tid]  = valid ? gate_list[e * T_TOKENS + r] : 0.0f;
  }
  __syncthreads();

#if HAVE_TDM
  // ---- TDM gather: one tensor_load_to_lds pulls the 16 gathered x rows ----
  if (wid == 0) {
    unsigned idx[16];
#pragma unroll
    for (int j = 0; j < 16; ++j) {
      int r = base + j;
      idx[j] = (r < count) ? (unsigned)tok_list[e * T_TOKENS + r] : 0u;
    }
    unsigned lds_off = (unsigned)(uintptr_t)(void*)xs;     // flat->LDS: addr[31:0]
    unsigned long long ga = (unsigned long long)(uintptr_t)xb;
    u32x4 g0;
    g0[0] = 1u | (1u << 31);                  // count=1, gather_mode=1, 16-bit idx
    g0[1] = lds_off;                          // LDS destination (bytes)
    g0[2] = (unsigned)(ga & 0xFFFFFFFFu);
    g0[3] = (unsigned)((ga >> 32) & 0x01FFFFFFu) | (2u << 30);   // addr[56:32], type=2
    i32x8 g1;
    g1[0] = (int)((1u << 16)                  // data_size = 2 bytes
                | (1u << 20)                  // pad_enable
                | (7u << 22)                  // pad_interval = 256 DWORDs (1024 B)
                | (3u << 25));                // pad_amount   = 4 DWORDs (16 B)
    g1[1] = (int)((unsigned)DM << 16);        // tensor_dim0[15:0]=1024
    g1[2] = (int)((unsigned)T_TOKENS << 16);  // tensor_dim0 hi=0 | tensor_dim1[15:0]=8192
    g1[3] = (int)((unsigned)DM << 16);        // tensor_dim1 hi=0 | tile_dim0=1024
    g1[4] = 16;                               // tile_dim1 = #valid gather indices
    g1[5] = DM;                               // tensor_dim0_stride = 1024 elems
    g1[6] = 0;
    g1[7] = 0;
    i32x4 g2, g3;
#pragma unroll
    for (int j = 0; j < 4; ++j) {
      g2[j] = __builtin_amdgcn_readfirstlane((int)(idx[2 * j]     | (idx[2 * j + 1] << 16)));
      g3[j] = __builtin_amdgcn_readfirstlane((int)(idx[8 + 2 * j] | (idx[9 + 2 * j] << 16)));
    }
    i32x8 g4 = {0, 0, 0, 0, 0, 0, 0, 0};      // unused descriptor slot (VADDR4/NULL)
    __builtin_amdgcn_tensor_load_to_lds(g0, g1, g2, g3, g4, 0);
    __builtin_amdgcn_s_wait_tensorcnt(0);
  }
  __syncthreads();
#else
  // ---- manual staging fallback (identical padded LDS layout) ----
  for (int i = tid; i < 16 * 128; i += 256) {
    int row = i >> 7, cv = i & 127;
    uint4 q = *((const uint4*)(xb + (size_t)tok[row] * DM) + cv);
    *(uint4*)(xs + row * XS_ROW + cv * 8 + ((cv >> 6) << 3)) = q;
  }
  __syncthreads();
#endif

  const int m  = lane & 15;   // A row / B,C column owned by this lane
  const int hf = lane >> 4;   // half selector

  // ---- GEMM1 + bias + exact GELU -> hs (bf16) ----
  const bf16_t* w1e = w1t + (size_t)e * DE * DM;
  const bf16_t* xrowp = xs + m * XS_ROW;
  for (int n1 = wid; n1 < DE / 16; n1 += 8) {
    v8f acc = {0.f, 0.f, 0.f, 0.f, 0.f, 0.f, 0.f, 0.f};
    const bf16_t* brow = w1e + (size_t)(n1 * 16 + m) * DM;
    if (n1 + 8 < DE / 16)
      __builtin_prefetch(w1e + (size_t)((n1 + 8) * 16 + m) * DM, 0, 0);
#pragma unroll 4
    for (int kk = 0; kk < DM / 32; ++kk) {
      int k0 = kk * 32;
      int ka = k0 + hf * 8;
      int kb = k0 + 16 + hf * 8;
      FragU a, b;
      a.q[0] = *(const uint4*)(xrowp + ka + ((ka >> 9) << 3));
      a.q[1] = *(const uint4*)(xrowp + kb + ((kb >> 9) << 3));
      b.q[0] = *(const uint4*)(brow + k0 + hf * 8);
      b.q[1] = *(const uint4*)(brow + k0 + 16 + hf * 8);
      acc = __builtin_amdgcn_wmma_f32_16x16x32_bf16(false, a.v, false, b.v,
                                                    (short)0, acc, false, false);
    }
    int ncol  = n1 * 16 + m;
    float b1v = b1[e * DE + ncol];
#pragma unroll
    for (int v = 0; v < 8; ++v) {
      float h = acc[v] + b1v;
      h = 0.5f * h * (1.0f + erff(h * 0.70710678118654752f));
      hs[(v + 8 * hf) * HS_STRIDE + ncol] = (bf16_t)h;
    }
  }
  __syncthreads();

  // ---- GEMM2 + bias, gate-weighted atomic scatter into out ----
  const bf16_t* w2e = w2t + (size_t)e * DM * DE;
  for (int n2 = wid; n2 < DM / 16; n2 += 8) {
    v8f acc = {0.f, 0.f, 0.f, 0.f, 0.f, 0.f, 0.f, 0.f};
    const bf16_t* brow = w2e + (size_t)(n2 * 16 + m) * DE;
    if (n2 + 8 < DM / 16)
      __builtin_prefetch(w2e + (size_t)((n2 + 8) * 16 + m) * DE, 0, 0);
#pragma unroll 4
    for (int kk = 0; kk < DE / 32; ++kk) {
      int k0 = kk * 32;
      FragU a, b;
      a.q[0] = *(const uint4*)(hs + m * HS_STRIDE + k0 + hf * 8);
      a.q[1] = *(const uint4*)(hs + m * HS_STRIDE + k0 + 16 + hf * 8);
      b.q[0] = *(const uint4*)(brow + k0 + hf * 8);
      b.q[1] = *(const uint4*)(brow + k0 + 16 + hf * 8);
      acc = __builtin_amdgcn_wmma_f32_16x16x32_bf16(false, a.v, false, b.v,
                                                    (short)0, acc, false, false);
    }
    int col   = n2 * 16 + m;
    float b2v = b2[e * DM + col];
#pragma unroll
    for (int v = 0; v < 8; ++v) {
      int row = v + 8 * hf;
      if (base + row < count) {
        atomicAdd(out + (size_t)tok[row] * DM + col, gv[row] * (acc[v] + b2v));
      }
    }
  }
}

// ---------------- launch ----------------
extern "C" void kernel_launch(void* const* d_in, const int* in_sizes, int n_in,
                              void* d_out, int out_size, void* d_ws, size_t ws_size,
                              hipStream_t stream) {
  const float* x    = (const float*)d_in[0];
  const float* w1   = (const float*)d_in[1];
  const float* b1   = (const float*)d_in[2];
  const float* w2   = (const float*)d_in[3];
  const float* b2   = (const float*)d_in[4];
  const float* cent = (const float*)d_in[5];
  const float* wr   = (const float*)d_in[6];
  float* out = (float*)d_out;

  char* ws = (char*)d_ws;
  int*    counts = (int*)(ws + WS_COUNTS);
  int*    tokl   = (int*)(ws + WS_TOK);
  float*  gatel  = (float*)(ws + WS_GATE);
  float*  cinv   = (float*)(ws + WS_CINV);
  bf16_t* xb     = (bf16_t*)(ws + WS_XB);
  bf16_t* w1t    = (bf16_t*)(ws + WS_W1T);
  bf16_t* w2t    = (bf16_t*)(ws + WS_W2T);

  const int n_out = T_TOKENS * DM;
  init_kernel  <<<n_out / 256, 256, 0, stream>>>(out, n_out, counts);
  prep_x_kernel<<<n_out / 256, 256, 0, stream>>>(x, xb, n_out);
  prep_w1_kernel<<<(NE * DE * DM) / 256, 256, 0, stream>>>(w1, w1t);
  prep_w2_kernel<<<(NE * DM * DE) / 256, 256, 0, stream>>>(w2, w2t);
  cnorm_kernel <<<1, 256, 0, stream>>>(cent, cinv);
  route_kernel <<<T_TOKENS / 8, 256, 0, stream>>>(x, cent, wr, cinv, counts, tokl, gatel);

  dim3 grid(T_TOKENS / 16, NE);
  moe_ffn_kernel<<<grid, 256, 0, stream>>>(xb, w1t, w2t, b1, b2, counts, tokl, gatel, out);
}
